// EnhancedSpatialAttention_65807488909380
// MI455X (gfx1250) — matrix operand order repaired and checked by reading, not compiled, not run on hardware
//
#include <hip/hip_runtime.h>
#include <math.h>

// ---------------------------------------------------------------------------
// EnhancedSpatialAttention for MI455X (gfx1250, wave32, WMMA, async-LDS)
// B=16, C=128, H=64, W=512, NUM_PARTS=4, pc=32
// ---------------------------------------------------------------------------

#define BQ  16
#define CQ  128
#define HQ  64
#define WQ  512
#define NP  4
#define PC  32
#define BP  (BQ*NP)   // 64

typedef __attribute__((ext_vector_type(16))) _Float16 v16h;
typedef __attribute__((ext_vector_type(8)))  float    v8f;

// workspace layout (float units)
#define OFF_HAVG 0u
#define OFF_HMAX (OFF_HAVG + BP*PC*HQ)              // 131072
#define OFF_WAVG (OFF_HMAX + BP*PC*HQ)              // 262144
#define OFF_WMAX (OFF_WAVG + BP*PC*WQ)              // 1310720
#define OFF_PH   (OFF_WMAX + BP*PC*WQ)              // 2359296
#define OFF_PWB  (OFF_PH   + BP*PC*HQ)              // 2490368  (f16 B-swizzled pw: BP*PC*WQ halfs)
// total = 2490368 + BP*PC*WQ/2 = 3014656 floats = 11.5 MB (must fit in d_ws)

__device__ inline float wave_sum32(float v) {
#pragma unroll
    for (int m = 16; m > 0; m >>= 1) v += __shfl_xor(v, m, 32);
    return v;
}
__device__ inline float wave_max32(float v) {
#pragma unroll
    for (int m = 16; m > 0; m >>= 1) v = fmaxf(v, __shfl_xor(v, m, 32));
    return v;
}

// ---------------------------------------------------------------------------
// Kernel 1: per-(bp,c) plane pooled stats. grid = BP*PC, 256 threads.
// ---------------------------------------------------------------------------
__global__ void k1_pool(const float* __restrict__ x, float* __restrict__ ws) {
    const int blk  = blockIdx.x;          // bp*PC + c
    const int tid  = threadIdx.x;
    const int lane = tid & 31;
    const int wv   = tid >> 5;
    const size_t base = (size_t)blk * (HQ * WQ);

    float* havg = ws + OFF_HAVG + (size_t)blk * HQ;
    float* hmax = ws + OFF_HMAX + (size_t)blk * HQ;
    float* wavg = ws + OFF_WAVG + (size_t)blk * WQ;
    float* wmax = ws + OFF_WMAX + (size_t)blk * WQ;

    // rows: wave wv covers h = wv*8 .. wv*8+7, float4 loads
#pragma unroll
    for (int r = 0; r < 8; ++r) {
        const int h = wv * 8 + r;
        const float4* row = (const float4*)(x + base + (size_t)h * WQ);
        float s = 0.0f, mx = -INFINITY;
#pragma unroll
        for (int k = 0; k < 4; ++k) {
            float4 v = row[lane + k * 32];
            s += (v.x + v.y) + (v.z + v.w);
            mx = fmaxf(mx, fmaxf(fmaxf(v.x, v.y), fmaxf(v.z, v.w)));
        }
        s  = wave_sum32(s);
        mx = wave_max32(mx);
        if (lane == 0) { havg[h] = s * (1.0f / WQ); hmax[h] = mx; }
    }

    // columns: thread owns cols 2*tid, 2*tid+1 (float2 rows, fully coalesced)
    float s0 = 0.0f, s1 = 0.0f, m0 = -INFINITY, m1 = -INFINITY;
    for (int h = 0; h < HQ; ++h) {
        float2 v = ((const float2*)(x + base + (size_t)h * WQ))[tid];
        s0 += v.x; m0 = fmaxf(m0, v.x);
        s1 += v.y; m1 = fmaxf(m1, v.y);
    }
    wavg[2 * tid]     = s0 * (1.0f / HQ); wmax[2 * tid]     = m0;
    wavg[2 * tid + 1] = s1 * (1.0f / HQ); wmax[2 * tid + 1] = m1;
}

// ---------------------------------------------------------------------------
// Kernel 2a: ph conv (3x2, pad H=1) + bias + relu + softmax over H.
// ---------------------------------------------------------------------------
__global__ void k2_ph(const float* __restrict__ ws_in,
                      const float* __restrict__ w_h, const float* __restrict__ b_h,
                      float* __restrict__ ws) {
    __shared__ float havg_s[PC][HQ];
    __shared__ float hmax_s[PC][HQ];
    __shared__ float ph_s[PC][HQ];

    const int bp  = blockIdx.x;
    const int tid = threadIdx.x;

#pragma unroll
    for (int i = 0; i < (PC * HQ) / 256; ++i) {
        int idx = tid + i * 256;
        int c = idx / HQ, h = idx % HQ;
        havg_s[c][h] = ws_in[OFF_HAVG + ((size_t)bp * PC + c) * HQ + h];
        hmax_s[c][h] = ws_in[OFF_HMAX + ((size_t)bp * PC + c) * HQ + h];
    }
    __syncthreads();

#pragma unroll
    for (int i = 0; i < 8; ++i) {
        int idx = tid * 8 + i;
        int o = idx / HQ, h = idx % HQ;
        float acc = b_h[o];
        for (int c = 0; c < PC; ++c) {
#pragma unroll
            for (int kh = 0; kh < 3; ++kh) {
                int hh = h + kh - 1;
                if (hh < 0 || hh >= HQ) continue;
                const float* wp = w_h + (((size_t)o * PC + c) * 3 + kh) * 2;
                acc += wp[0] * havg_s[c][hh] + wp[1] * hmax_s[c][hh];
            }
        }
        ph_s[o][h] = fmaxf(acc, 0.0f);
    }
    __syncthreads();

    if (tid < PC) {
        const int o = tid;
        float mx = -INFINITY;
        for (int h = 0; h < HQ; ++h) mx = fmaxf(mx, ph_s[o][h]);
        float s = 0.0f;
        for (int h = 0; h < HQ; ++h) { float e = __expf(ph_s[o][h] - mx); ph_s[o][h] = e; s += e; }
        float inv = 1.0f / s;
        float* out = ws + OFF_PH + ((size_t)bp * PC + o) * HQ;
        for (int h = 0; h < HQ; ++h) out[h] = ph_s[o][h] * inv;
    }
}

// ---------------------------------------------------------------------------
// Kernel 2b: pw conv (2x3, pad W=1) + bias + relu + softmax over W.
// Emits pw as f16 in WMMA B-register-swizzled layout:
//   element (k=row, n=w) -> [ (w>>4)*32 + (w&15) + 16*((k>>3)&1) ] fragment, half (k&7) + 8*(k>>4)
// so kernel 3 lanes load contiguous 32B fragments.
// ---------------------------------------------------------------------------
__global__ void k2_pw(const float* __restrict__ ws_in,
                      const float* __restrict__ w_w, const float* __restrict__ b_w,
                      float* __restrict__ ws) {
    __shared__ float pw_s[PC][WQ];             // 64 KB

    const int bp   = blockIdx.x;
    const int tid  = threadIdx.x;
    const int lane = tid & 31;
    const int wv   = tid >> 5;

    const float* wavg = ws_in + OFF_WAVG + (size_t)bp * PC * WQ;
    const float* wmax = ws_in + OFF_WMAX + (size_t)bp * PC * WQ;

    for (int i = 0; i < (PC * WQ) / 256; ++i) {
        int idx = tid + i * 256;
        int o = idx >> 9, w = idx & (WQ - 1);
        float acc = b_w[o];
        for (int c = 0; c < PC; ++c) {
            const float* wp = w_w + ((size_t)o * PC + c) * 6;
            const float* av = wavg + (size_t)c * WQ;
            const float* mv = wmax + (size_t)c * WQ;
#pragma unroll
            for (int kw = 0; kw < 3; ++kw) {
                int wwi = w + kw - 1;
                if (wwi < 0 || wwi >= WQ) continue;
                acc += wp[0 * 3 + kw] * av[wwi] + wp[1 * 3 + kw] * mv[wwi];
            }
        }
        pw_s[o][w] = fmaxf(acc, 0.0f);
    }
    __syncthreads();

    _Float16* pwb = (_Float16*)(ws + OFF_PWB) + (size_t)bp * (PC * WQ);

#pragma unroll
    for (int rr = 0; rr < 4; ++rr) {
        const int o = wv * 4 + rr;            // B-matrix k index
        float mx = -INFINITY;
#pragma unroll
        for (int k = 0; k < WQ / 32; ++k) mx = fmaxf(mx, pw_s[o][lane + k * 32]);
        mx = wave_max32(mx);
        float s = 0.0f;
#pragma unroll
        for (int k = 0; k < WQ / 32; ++k) s += __expf(pw_s[o][lane + k * 32] - mx);
        s = wave_sum32(s);
        const float inv = 1.0f / s;
        const int lane2 = (o & 8) << 1;       // +16 if bit3 of k
        const int e     = (o & 7) + ((o >> 4) << 3);
#pragma unroll
        for (int k = 0; k < WQ / 32; ++k) {
            int w = lane + k * 32;
            float v = __expf(pw_s[o][w] - mx) * inv;
            int frag = ((w >> 4) * 32) + (w & 15) + lane2;
            pwb[((size_t)frag << 4) + e] = (_Float16)v;
        }
    }
}

// ---------------------------------------------------------------------------
// Kernel 3: amap GEMM via v_wmma_f32_16x16x32_f16 + fused sigmoid gating.
// grid = BP*32 = 2048 blocks (bp, o). Block owns M=64 (all h of one o) x N=512.
//   - pw fragments (32 KB, pre-swizzled f16) pulled in via
//     global_load_async_to_lds_b128 (ASYNCcnt).
//   - U[h][c] = w_e[o][c]*ph[c][h] built in A-register-swizzled LDS.
//   - 8 waves x 4 ntiles x 4 mtiles = 128 WMMAs; B frag loaded once per ntile
//     (2x ds_load_b128) feeds 4 WMMAs; A frags hoisted to registers.
// ---------------------------------------------------------------------------
__global__ void k3_attend(const float* __restrict__ x,
                          const float* __restrict__ w_e, const float* __restrict__ b_e,
                          const float* __restrict__ ws,
                          float* __restrict__ out) {
    __shared__ v16h pwB_lds[32 * 32];          // [ntile][lane] fragments, 32 KB
    __shared__ v16h uA_lds[4 * 32];            // [mtile][lane] fragments, 4 KB

    const int tid  = threadIdx.x;
    const int lane = tid & 31;
    const int wv   = tid >> 5;

    const int bp   = blockIdx.x >> 5;
    const int o    = blockIdx.x & 31;          // output channel of the part
    const int b    = bp / NP;
    const int part = bp % NP;

    // ---- async copy of pre-swizzled pw fragments: 32 KB = 2048 x b128
    {
        const _Float16* src = (const _Float16*)(ws + OFF_PWB) + (size_t)bp * (PC * WQ);
        unsigned ldsb = (unsigned)(uintptr_t)&pwB_lds[0];
#pragma unroll
        for (int i = 0; i < 8; ++i) {
            unsigned boff = (unsigned)(tid + i * 256) * 16u;
            asm volatile("global_load_async_to_lds_b128 %0, %1, %2"
                         :: "v"(ldsb + boff), "v"(boff), "s"(src)
                         : "memory");
        }
    }

    // ---- build U (64 h x 32 c) into A-register-swizzled LDS
    // A frag layout: lane = rloc + 16*((c>>3)&1), half e = (c&7) + 8*(c>>4)
    {
        const float* php = ws + OFF_PH + (size_t)bp * PC * HQ;
        _Float16* ua = (_Float16*)&uA_lds[0];
#pragma unroll
        for (int i = 0; i < (HQ * PC) / 256; ++i) {   // 8 iters
            int idx = tid + i * 256;
            int h = idx >> 5, c = idx & 31;
            float u = w_e[(size_t)o * PC + c] * php[(size_t)c * HQ + h];
            int mt = h >> 4, rloc = h & 15;
            int la = rloc + ((c & 8) << 1);           // +16 if bit3 of c
            int e  = (c & 7) + ((c >> 4) << 3);       // +8 if c>=16
            ua[(((mt << 5) + la) << 4) + e] = (_Float16)u;
        }
    }

    asm volatile("s_wait_asynccnt 0x0" ::: "memory");
    __syncthreads();

    // ---- hoist the 4 A fragments (one per mtile) into registers
    v16h amat[4];
#pragma unroll
    for (int mt = 0; mt < 4; ++mt) amat[mt] = uA_lds[(mt << 5) + lane];

    const int   hi     = lane >> 4;
    const int   ncol   = lane & 15;
    const float be_o   = b_e[o];
    const size_t base_g = (((size_t)b * CQ + part * PC + o) * HQ) * WQ;

#pragma unroll
    for (int t = 0; t < 4; ++t) {
        const int ntile = wv * 4 + t;
        const int n0    = ntile << 4;
        v16h bmat = pwB_lds[(ntile << 5) + lane];     // 2x ds_load_b128

#pragma unroll
        for (int mt = 0; mt < 4; ++mt) {
            v8f acc = {};
            acc = __builtin_amdgcn_wmma_f32_16x16x32_f16(
                false, amat[mt], false, bmat, (short)0, acc, false, false);

            // D: VGPR r, lane l -> (m = r + 8*(l>=16), n = l&15)
#pragma unroll
            for (int r = 0; r < 8; ++r) {
                const int h = (mt << 4) + (hi << 3) + r;
                const int w = n0 + ncol;
                const size_t g = base_g + (size_t)h * WQ + w;
                const float amap = acc[r] + be_o;
                const float sg = 1.0f / (1.0f + __expf(-amap));
                const float xv = __builtin_nontemporal_load(x + g);
                __builtin_nontemporal_store(xv * (1.0f + sg), out + g);
            }
        }
    }
}

// ---------------------------------------------------------------------------
extern "C" void kernel_launch(void* const* d_in, const int* in_sizes, int n_in,
                              void* d_out, int out_size, void* d_ws, size_t ws_size,
                              hipStream_t stream) {
    const float* x   = (const float*)d_in[0];
    const float* w_h = (const float*)d_in[1];
    const float* b_h = (const float*)d_in[2];
    const float* w_w = (const float*)d_in[3];
    const float* b_w = (const float*)d_in[4];
    const float* w_e = (const float*)d_in[5];
    const float* b_e = (const float*)d_in[6];
    float*       out = (float*)d_out;
    float*       ws  = (float*)d_ws;   // needs ~11.5 MB

    k1_pool  <<<BP * PC, 256, 0, stream>>>(x, ws);
    k2_ph    <<<BP,      256, 0, stream>>>(ws, w_h, b_h, ws);
    k2_pw    <<<BP,      256, 0, stream>>>(ws, w_w, b_w, ws);
    k3_attend<<<BP * PC, 256, 0, stream>>>(x, w_e, b_e, ws, out);

    (void)in_sizes; (void)n_in; (void)out_size; (void)ws_size;
}
